// Mamba2Block_9715216023902
// MI455X (gfx1250) — compile-verified, matrix operand order checked by reading, not run on hardware
//
#include <hip/hip_runtime.h>
#include <hip/hip_bf16.h>

// ---------------------------------------------------------------------------
// Mamba2 block forward for MI455X (gfx1250).
// All GEMM-shaped work -> v_wmma_f32_16x16x32_bf16 (fp32 accum).
// bf16 operand tiles staged global->LDS via async-to-LDS when available.
// ---------------------------------------------------------------------------

typedef __attribute__((ext_vector_type(16))) __bf16 bf16x16;
typedef __attribute__((ext_vector_type(8)))  float  f32x8;

#define WMMA_BF16(a, b, c) \
  __builtin_amdgcn_wmma_f32_16x16x32_bf16(false, (a), false, (b), (short)0, (c), false, false)

// Model constants (from the reference)
constexpr int kB    = 2;
constexpr int kL    = 2048;
constexpr int kBL   = kB * kL;        // 4096 token rows
constexpr int kDM   = 1024;           // d_model
constexpr int kDI   = 2048;           // d_inner
constexpr int kDIP  = 4368;           // d_in_proj
constexpr int kCONV = 2304;           // conv_dim
constexpr int kH    = 16;             // heads
constexpr int kP    = 128;            // head_dim
constexpr int kN    = 128;            // d_state
constexpr int kCS   = 256;            // chunk size
constexpr int kNC   = kL / kCS;       // 8 chunks

// ---------------- bf16 helpers ----------------
__device__ __forceinline__ unsigned short f2bf(float f) {
  unsigned int u = __builtin_bit_cast(unsigned int, f);
  u += 0x7fffu + ((u >> 16) & 1u);      // round-to-nearest-even
  return (unsigned short)(u >> 16);
}
__device__ __forceinline__ float bf2f(unsigned short h) {
  unsigned int u = ((unsigned int)h) << 16;
  return __builtin_bit_cast(float, u);
}
__device__ __forceinline__ f32x8 zero8() {
  f32x8 z;
#pragma unroll
  for (int i = 0; i < 8; ++i) z[i] = 0.f;
  return z;
}

// ---------------- async global->LDS copy (gfx1250 path, guarded) ----------
#if defined(__has_builtin)
#if __has_builtin(__builtin_amdgcn_global_load_async_to_lds_b128) && \
    __has_builtin(__builtin_amdgcn_s_wait_asynccnt)
#define HAVE_ASYNC_COPY 1
#endif
#endif
#ifndef HAVE_ASYNC_COPY
#define HAVE_ASYNC_COPY 0
#endif

#if HAVE_ASYNC_COPY
typedef int v4i_async __attribute__((vector_size(16)));
typedef __attribute__((address_space(1))) v4i_async* gptr_v4i;
typedef __attribute__((address_space(3))) v4i_async* lptr_v4i;
#endif

__device__ __forceinline__ void copy16_to_lds(const unsigned short* g, unsigned short* l) {
#if HAVE_ASYNC_COPY
  __builtin_amdgcn_global_load_async_to_lds_b128(
      (gptr_v4i)(void*)g, (lptr_v4i)(void*)l, 0, 0);
#else
  *(uint4*)l = *(const uint4*)g;
#endif
}
__device__ __forceinline__ void async_fence() {
#if HAVE_ASYNC_COPY
  __builtin_amdgcn_s_wait_asynccnt(0);
#endif
}

// A-operand (or [N][K]-stored B-operand) fragment: 16 rows x 32 K, bf16.
// lanes 0-15: row = lane, K in {0..7,16..23}; lanes 16-31: row = lane-16, K+8.
__device__ __forceinline__ bf16x16 frag_row(const unsigned short* base, int ld, int lane) {
  const unsigned short* p = base + (lane & 15) * ld + ((lane >> 4) * 8);
  bf16x16 v;
#pragma unroll
  for (int e = 0; e < 16; ++e) {
    int pr = e >> 1;
    int k  = ((pr < 4) ? (pr * 2) : (pr * 2 + 8)) + (e & 1);
    v[e] = __builtin_bit_cast(__bf16, p[k]);
  }
  return v;
}
// B-operand fragment when storage is [K][N] row-major: column n = lane&15.
__device__ __forceinline__ bf16x16 frag_col(const unsigned short* base, int ld, int lane) {
  const unsigned short* p = base + (lane & 15) + ((lane >> 4) * 8) * ld;
  bf16x16 v;
#pragma unroll
  for (int e = 0; e < 16; ++e) {
    int pr = e >> 1;
    int k  = ((pr < 4) ? (pr * 2) : (pr * 2 + 8)) + (e & 1);
    v[e] = __builtin_bit_cast(__bf16, p[k * ld]);
  }
  return v;
}

// ---------------------------------------------------------------------------
// Generic GEMM: C[M,N] = A[M,K] * W[N,K]^T (+ bias), f32 in/out, bf16 WMMA.
// Block tile 128(M) x 64(N), K step 64 (2 WMMA K-steps per barrier pair).
// Edge tiles handled by row clamping (loads) + store guards: no exec branches.
// ---------------------------------------------------------------------------
__global__ __launch_bounds__(256) void k_gemm(const float* __restrict__ A,
                                              const float* __restrict__ W,
                                              const float* __restrict__ bias,
                                              float* __restrict__ C,
                                              int M, int N, int K) {
  constexpr int GBK = 64;
  __shared__ unsigned short sA[128 * GBK];   // 16 KB
  __shared__ unsigned short sW[64 * GBK];    //  8 KB
  const int tid = threadIdx.x, lane = tid & 31, wave = tid >> 5;
  const int bm = blockIdx.y * 128, bn = blockIdx.x * 64;
  const int wm = (wave & 3) * 32, wn = (wave >> 2) * 32;

  f32x8 acc[2][2];
#pragma unroll
  for (int i = 0; i < 2; ++i)
#pragma unroll
    for (int j = 0; j < 2; ++j) acc[i][j] = zero8();

  for (int k0 = 0; k0 < K; k0 += GBK) {
    __syncthreads();
    if (k0 + GBK < K) {   // prefetch next K tile into cache (global_prefetch_b8)
      int pr = min(bm + (tid >> 1), M - 1);
      __builtin_prefetch(A + (size_t)pr * K + k0 + GBK + (tid & 1) * 32, 0, 0);
      int pw = min(bn + (tid >> 2), N - 1);
      __builtin_prefetch(W + (size_t)pw * K + k0 + GBK + (tid & 3) * 16, 0, 0);
    }
#pragma unroll
    for (int it = 0; it < 8; ++it) {          // A tile: 128x64 floats
      int idx = (tid + it * 256) * 4;
      int r = idx >> 6, c = idx & 63;
      int gr = min(bm + r, M - 1);
      float4 v = *(const float4*)(A + (size_t)gr * K + k0 + c);
      sA[r * GBK + c + 0] = f2bf(v.x);
      sA[r * GBK + c + 1] = f2bf(v.y);
      sA[r * GBK + c + 2] = f2bf(v.z);
      sA[r * GBK + c + 3] = f2bf(v.w);
    }
#pragma unroll
    for (int it = 0; it < 4; ++it) {          // W tile: 64x64 floats
      int idx = (tid + it * 256) * 4;
      int r = idx >> 6, c = idx & 63;
      int gr = min(bn + r, N - 1);
      float4 v = *(const float4*)(W + (size_t)gr * K + k0 + c);
      sW[r * GBK + c + 0] = f2bf(v.x);
      sW[r * GBK + c + 1] = f2bf(v.y);
      sW[r * GBK + c + 2] = f2bf(v.z);
      sW[r * GBK + c + 3] = f2bf(v.w);
    }
    __syncthreads();

#pragma unroll
    for (int kk = 0; kk < GBK; kk += 32) {
      bf16x16 a0 = frag_row(sA + (wm + 0)  * GBK + kk, GBK, lane);
      bf16x16 a1 = frag_row(sA + (wm + 16) * GBK + kk, GBK, lane);
      bf16x16 b0 = frag_row(sW + (wn + 0)  * GBK + kk, GBK, lane);
      bf16x16 b1 = frag_row(sW + (wn + 16) * GBK + kk, GBK, lane);
      acc[0][0] = WMMA_BF16(a0, b0, acc[0][0]);
      acc[0][1] = WMMA_BF16(a0, b1, acc[0][1]);
      acc[1][0] = WMMA_BF16(a1, b0, acc[1][0]);
      acc[1][1] = WMMA_BF16(a1, b1, acc[1][1]);
    }
  }

  const int half = lane >> 4, ln = lane & 15;
#pragma unroll
  for (int mi = 0; mi < 2; ++mi)
#pragma unroll
    for (int ni = 0; ni < 2; ++ni) {
      int col = bn + wn + ni * 16 + ln;
      if (col >= N) continue;
      float badd = bias ? bias[col] : 0.f;
#pragma unroll
      for (int r = 0; r < 8; ++r) {
        int row = bm + wm + mi * 16 + half * 8 + r;
        if (row < M) C[(size_t)row * N + col] = acc[mi][ni][r] + badd;
      }
    }
}

// ---------------------------------------------------------------------------
// dt softplus + dA = dt * (-exp(A_log))
// ---------------------------------------------------------------------------
__global__ void k_dtprep(const float* __restrict__ zx, const float* __restrict__ dt_bias,
                         const float* __restrict__ A_log,
                         float* __restrict__ dtsp, float* __restrict__ dA) {
  int i = blockIdx.x * 256 + threadIdx.x;      // over kBL*kH
  if (i >= kBL * kH) return;
  int bl = i >> 4, h = i & 15;
  float x  = zx[(size_t)bl * kDIP + (kDI + kCONV) + h] + dt_bias[h];
  float sp = (x > 20.f) ? x : log1pf(__expf(x));
  dtsp[i] = sp;
  int b = bl / kL, l = bl % kL;
  dA[((size_t)(b * kH + h)) * kL + l] = sp * (-__expf(A_log[h]));
}

// ---------------------------------------------------------------------------
// Depthwise causal conv (width 4) + bias + SiLU; route outputs:
//   ch <  kDI        -> x (f32) and x*dt (bf16)
//   ch <  kDI+kN     -> B (bf16)
//   else             -> C (bf16)
// ---------------------------------------------------------------------------
__global__ void k_conv(const float* __restrict__ zx, const float* __restrict__ conv_w,
                       const float* __restrict__ conv_b, const float* __restrict__ dtsp,
                       float* __restrict__ xf, unsigned short* __restrict__ xdt,
                       unsigned short* __restrict__ Bbf, unsigned short* __restrict__ Cbf) {
  size_t i = (size_t)blockIdx.x * 256 + threadIdx.x;
  if (i >= (size_t)kBL * kCONV) return;
  int ch = (int)(i % kCONV);
  size_t bl = i / kCONV;
  int b = (int)(bl / kL), l = (int)(bl % kL);
  float s = conv_b[ch];
#pragma unroll
  for (int j = 0; j < 4; ++j) {
    int ls = l - 3 + j;
    if (ls >= 0) s += zx[((size_t)(b * kL + ls)) * kDIP + kDI + ch] * conv_w[ch * 4 + j];
  }
  float v = s / (1.f + __expf(-s));            // SiLU
  if (ch < kDI) {
    int h = ch >> 7;
    xf[bl * kDI + ch]  = v;
    xdt[bl * kDI + ch] = f2bf(v * dtsp[bl * kH + h]);
  } else if (ch < kDI + kN) {
    Bbf[bl * kN + (ch - kDI)] = f2bf(v);
  } else {
    Cbf[bl * kN + (ch - kDI - kN)] = f2bf(v);
  }
}

// ---------------------------------------------------------------------------
// Per-chunk inclusive cumsum of dA. One block = one (b,c,h) chunk of 256.
// ---------------------------------------------------------------------------
__global__ void k_cumsum(const float* __restrict__ dA, float* __restrict__ acum) {
  __shared__ float s[256];
  int x = blockIdx.x;                       // b*128 + c*16 + h
  int h = x & 15, c = (x >> 4) & 7, b = x >> 7;
  int i = threadIdx.x;
  s[i] = dA[((size_t)(b * kH + h)) * kL + c * kCS + i];
  __syncthreads();
  for (int o = 1; o < 256; o <<= 1) {
    float t = 0.f;
    if (i >= o) t = s[i - o];
    __syncthreads();
    s[i] += t;
    __syncthreads();
  }
  acum[(size_t)x * kCS + i] = s[i];
}

// ---------------------------------------------------------------------------
// Y_diag per (b,c,h): Y[i,p] = sum_{j<=i} exp(ac_i-ac_j) * (C_i . B_j) * xdt_j[p]
// CB via WMMA in 64-wide j tiles; masked/decayed in regs; M*X via WMMA.
// Each wave owns a 32-row i strip, full 128 p columns (16 persistent accums).
// ---------------------------------------------------------------------------
__global__ __launch_bounds__(256) void k_ydiag(const unsigned short* __restrict__ Cbf,
                                               const unsigned short* __restrict__ Bbf,
                                               const unsigned short* __restrict__ xdt,
                                               const float* __restrict__ acum,
                                               float* __restrict__ y) {
  extern __shared__ char smem[];
  unsigned short* sC  = (unsigned short*)smem;        // 256x128 [i][n]
  unsigned short* sBt = sC + kCS * kN;                // 64x128  [j][n]
  unsigned short* sXt = sBt + 64 * kN;                // 64x128  [j][p]
  unsigned short* sM  = sXt + 64 * kP;                // 8 waves * 32x64
  float* sAc = (float*)(sM + 8 * 32 * 64);            // 256

  const int tid = threadIdx.x, lane = tid & 31, wave = tid >> 5;
  const int x = blockIdx.x;
  const int h = x & 15, c = (x >> 4) & 7, b = x >> 7;
  const size_t rowbase = (size_t)(b * kL + c * kCS);
  const int half = lane >> 4, ln = lane & 15;
  const int iw = wave * 32;

  // C tile (bf16, byte copy) -> async path when available
  for (int t = tid; t < (kCS * kN) / 8; t += 256) {
    int row = t >> 4, c8 = (t & 15) * 8;
    copy16_to_lds(&Cbf[(rowbase + row) * kN + c8], &sC[row * kN + c8]);
  }
  for (int i = tid; i < kCS; i += 256) sAc[i] = acum[(size_t)x * kCS + i];
  async_fence();

  unsigned short* sMw = sM + wave * (32 * 64);
  f32x8 yacc[2][8];
#pragma unroll
  for (int mi = 0; mi < 2; ++mi)
#pragma unroll
    for (int pi = 0; pi < 8; ++pi) yacc[mi][pi] = zero8();

  for (int jt = 0; jt < kCS; jt += 64) {
    __syncthreads();
    for (int t = tid; t < (64 * kN) / 8; t += 256) {
      int j = t >> 4, c8 = (t & 15) * 8;
      copy16_to_lds(&Bbf[(rowbase + jt + j) * kN + c8], &sBt[j * kN + c8]);
      copy16_to_lds(&xdt[(rowbase + jt + j) * kDI + h * kP + c8], &sXt[j * kN + c8]);
    }
    async_fence();
    __syncthreads();

    // CB tile (32 x 64 per wave), mask + decay, requantize to bf16
#pragma unroll
    for (int mi = 0; mi < 2; ++mi) {
#pragma unroll
      for (int nj = 0; nj < 4; ++nj) {
        f32x8 acc = zero8();
#pragma unroll
        for (int kk = 0; kk < kN; kk += 32) {
          bf16x16 a  = frag_row(sC + (iw + mi * 16) * kN + kk, kN, lane);
          bf16x16 bb = frag_row(sBt + (nj * 16) * kN + kk, kN, lane);
          acc = WMMA_BF16(a, bb, acc);
        }
        int jl = nj * 16 + ln;
        int jg = jt + jl;
        float aj = sAc[jg];
        int i0 = iw + mi * 16 + half * 8;
#pragma unroll
        for (int r = 0; r < 8; ++r) {
          int ig = i0 + r;
          float m = (jg <= ig) ? acc[r] * __expf(sAc[ig] - aj) : 0.f;
          sMw[(mi * 16 + half * 8 + r) * 64 + jl] = f2bf(m);
        }
      }
    }
    // Y += M (32x64) * X (64x128). sMw is private per wave; same-wave LDS is ordered.
#pragma unroll
    for (int mi = 0; mi < 2; ++mi)
#pragma unroll
      for (int pi = 0; pi < 8; ++pi) {
        f32x8 acc = yacc[mi][pi];
#pragma unroll
        for (int kk = 0; kk < 64; kk += 32) {
          bf16x16 a  = frag_row(sMw + (mi * 16) * 64 + kk, 64, lane);
          bf16x16 bb = frag_col(sXt + kk * kP + pi * 16, kP, lane);
          acc = WMMA_BF16(a, bb, acc);
        }
        yacc[mi][pi] = acc;
      }
  }

#pragma unroll
  for (int mi = 0; mi < 2; ++mi)
#pragma unroll
    for (int pi = 0; pi < 8; ++pi) {
      int col = h * kP + pi * 16 + ln;
#pragma unroll
      for (int r = 0; r < 8; ++r) {
        int row = iw + mi * 16 + half * 8 + r;
        y[(rowbase + row) * kDI + col] = yacc[mi][pi][r];
      }
    }
}

// ---------------------------------------------------------------------------
// Per-chunk states: S[p,n] = sum_j exp(ac_last - ac_j) * xdt_j[p] * B_j[n]
// ---------------------------------------------------------------------------
__global__ __launch_bounds__(256) void k_states(const unsigned short* __restrict__ Bbf,
                                                const unsigned short* __restrict__ xdt,
                                                const float* __restrict__ acum,
                                                float* __restrict__ states) {
  extern __shared__ char smem[];
  unsigned short* sB  = (unsigned short*)smem;   // 256x128 [j][n]
  unsigned short* sXT = sB + kCS * kN;           // 128x256 [p][j]

  const int tid = threadIdx.x, lane = tid & 31, wave = tid >> 5;
  const int x = blockIdx.x;
  const int h = x & 15, c = (x >> 4) & 7, b = x >> 7;
  const size_t rowbase = (size_t)(b * kL + c * kCS);
  const float alast = acum[(size_t)x * kCS + (kCS - 1)];

  // B tile: raw byte copy -> async path
  for (int t = tid; t < (kCS * kN) / 8; t += 256) {
    int j = t >> 4, c8 = (t & 15) * 8;
    copy16_to_lds(&Bbf[(rowbase + j) * kN + c8], &sB[j * kN + c8]);
  }
  // decayed x^T tile: needs VALU, plain path
  for (int i = tid; i < kCS * kN; i += 256) {
    int j = i >> 7, n = i & 127;
    float xv = bf2f(xdt[(rowbase + j) * kDI + h * kP + n]);   // n doubles as p
    float d  = __expf(alast - acum[(size_t)x * kCS + j]);
    sXT[n * kCS + j] = f2bf(xv * d);
  }
  async_fence();
  __syncthreads();

  f32x8 acc[8];
#pragma unroll
  for (int ni = 0; ni < 8; ++ni) acc[ni] = zero8();
  for (int kk = 0; kk < kCS; kk += 32) {
    bf16x16 a = frag_row(sXT + (wave * 16) * kCS + kk, kCS, lane);
#pragma unroll
    for (int ni = 0; ni < 8; ++ni) {
      bf16x16 bb = frag_col(sB + kk * kN + ni * 16, kN, lane);
      acc[ni] = WMMA_BF16(a, bb, acc[ni]);
    }
  }
  const int half = lane >> 4, ln = lane & 15;
#pragma unroll
  for (int ni = 0; ni < 8; ++ni) {
    int n = ni * 16 + ln;
#pragma unroll
    for (int r = 0; r < 8; ++r) {
      int p = wave * 16 + half * 8 + r;
      states[(size_t)x * (kP * kN) + (size_t)p * kN + n] = acc[ni][r];
    }
  }
}

// ---------------------------------------------------------------------------
// Inter-chunk exclusive scan: prior_c = carry; carry = carry*dec_c + states_c
// One block per (b,h); 8 sequential chunks; 64 state elems/thread in regs.
// ---------------------------------------------------------------------------
__global__ void k_scan(const float* __restrict__ states, const float* __restrict__ acum,
                       float* __restrict__ prior) {
  int x = blockIdx.x;                  // b*kH + h
  int b = x >> 4, h = x & 15;
  int tid = threadIdx.x;
  float run[64];
#pragma unroll
  for (int e = 0; e < 64; ++e) run[e] = 0.f;
  for (int c = 0; c < kNC; ++c) {
    size_t idx = (size_t)((b * kNC + c) * kH + h);
    float dec = __expf(acum[idx * kCS + (kCS - 1)]);
    size_t base = idx * (size_t)(kP * kN) + (size_t)tid * 64;
#pragma unroll
    for (int e = 0; e < 64; ++e) {
      prior[base + e] = run[e];
      run[e] = run[e] * dec + states[base + e];
    }
  }
}

// ---------------------------------------------------------------------------
// Y_off per (b,c,h): y += exp(ac_i) * (C_i . prior[p,:]) + D[h]*x  (in place)
// ---------------------------------------------------------------------------
__global__ __launch_bounds__(256) void k_yoff(const unsigned short* __restrict__ Cbf,
                                              const float* __restrict__ prior,
                                              const float* __restrict__ acum,
                                              const float* __restrict__ xf,
                                              const float* __restrict__ D_param,
                                              float* __restrict__ y) {
  extern __shared__ char smem[];
  unsigned short* sC = (unsigned short*)smem;    // 256x128 [i][n]
  unsigned short* sP = sC + kCS * kN;            // 128x128 [p][n]
  float* sAc = (float*)(sP + kP * kN);           // 256

  const int tid = threadIdx.x, lane = tid & 31, wave = tid >> 5;
  const int x = blockIdx.x;
  const int h = x & 15, c = (x >> 4) & 7, b = x >> 7;
  const size_t rowbase = (size_t)(b * kL + c * kCS);
  const int half = lane >> 4, ln = lane & 15;
  const int iw = wave * 32;

  for (int t = tid; t < (kCS * kN) / 8; t += 256) {
    int row = t >> 4, c8 = (t & 15) * 8;
    copy16_to_lds(&Cbf[(rowbase + row) * kN + c8], &sC[row * kN + c8]);
  }
  for (int i = tid; i < kP * kN; i += 256)
    sP[i] = f2bf(prior[(size_t)x * (kP * kN) + i]);
  for (int i = tid; i < kCS; i += 256) sAc[i] = acum[(size_t)x * kCS + i];
  async_fence();
  __syncthreads();

  f32x8 acc[2][8];
#pragma unroll
  for (int mi = 0; mi < 2; ++mi)
#pragma unroll
    for (int pi = 0; pi < 8; ++pi) acc[mi][pi] = zero8();

  for (int kk = 0; kk < kN; kk += 32) {
    bf16x16 a0 = frag_row(sC + (iw + 0)  * kN + kk, kN, lane);
    bf16x16 a1 = frag_row(sC + (iw + 16) * kN + kk, kN, lane);
#pragma unroll
    for (int pi = 0; pi < 8; ++pi) {
      bf16x16 bb = frag_row(sP + (pi * 16) * kN + kk, kN, lane);  // row p contiguous in n
      acc[0][pi] = WMMA_BF16(a0, bb, acc[0][pi]);
      acc[1][pi] = WMMA_BF16(a1, bb, acc[1][pi]);
    }
  }

  const float dpar = D_param[h];
#pragma unroll
  for (int mi = 0; mi < 2; ++mi)
#pragma unroll
    for (int pi = 0; pi < 8; ++pi) {
      int col = h * kP + pi * 16 + ln;
#pragma unroll
      for (int r = 0; r < 8; ++r) {
        int row = iw + mi * 16 + half * 8 + r;
        size_t gi = (rowbase + row) * kDI + col;
        y[gi] = y[gi] + __expf(sAc[row]) * acc[mi][pi][r] + dpar * xf[gi];
      }
    }
}

// ---------------------------------------------------------------------------
// y *= silu(z); RMS-style scale across d_inner; * norm_w  -> yn
// ---------------------------------------------------------------------------
__global__ __launch_bounds__(256) void k_gatenorm(const float* __restrict__ y,
                                                  const float* __restrict__ zx,
                                                  const float* __restrict__ norm_w,
                                                  float* __restrict__ yn) {
  __shared__ float red[256];
  const int bl = blockIdx.x, tid = threadIdx.x;
  float v[8];
  float ss = 0.f;
#pragma unroll
  for (int e = 0; e < 8; ++e) {
    int d = tid + e * 256;
    float yy = y[(size_t)bl * kDI + d];
    float z  = zx[(size_t)bl * kDIP + d];
    float g  = yy * (z / (1.f + __expf(-z)));
    v[e] = g;
    ss += g * g;
  }
  red[tid] = ss;
  __syncthreads();
  for (int o = 128; o > 0; o >>= 1) {
    if (tid < o) red[tid] += red[tid + o];
    __syncthreads();
  }
  float scale = rsqrtf(red[0] / (float)kDI + 1e-5f);
#pragma unroll
  for (int e = 0; e < 8; ++e) {
    int d = tid + e * 256;
    yn[(size_t)bl * kDI + d] = v[e] * scale * norm_w[d];
  }
}

// ---------------------------------------------------------------------------
// Workspace layout (bytes)
// ---------------------------------------------------------------------------
constexpr size_t SZ_Z    = (size_t)kBL * kDIP * 4;
constexpr size_t SZ_DTSP = (size_t)kBL * kH * 4;
constexpr size_t SZ_DA   = (size_t)kB * kH * kL * 4;
constexpr size_t SZ_ACUM = (size_t)kB * kNC * kH * kCS * 4;
constexpr size_t SZ_X    = (size_t)kBL * kDI * 4;
constexpr size_t SZ_XDT  = (size_t)kBL * kDI * 2;
constexpr size_t SZ_BBF  = (size_t)kBL * kN * 2;
constexpr size_t SZ_CBF  = (size_t)kBL * kN * 2;
constexpr size_t SZ_Y    = (size_t)kBL * kDI * 4;
constexpr size_t SZ_ST   = (size_t)kB * kNC * kH * kP * kN * 4;

constexpr size_t OFF_Z    = 0;
constexpr size_t OFF_DTSP = OFF_Z + SZ_Z;
constexpr size_t OFF_DA   = OFF_DTSP + SZ_DTSP;
constexpr size_t OFF_ACUM = OFF_DA + SZ_DA;
constexpr size_t OFF_X    = OFF_ACUM + SZ_ACUM;
constexpr size_t OFF_XDT  = OFF_X + SZ_X;
constexpr size_t OFF_BBF  = OFF_XDT + SZ_XDT;
constexpr size_t OFF_CBF  = OFF_BBF + SZ_BBF;
constexpr size_t OFF_Y    = OFF_CBF + SZ_CBF;
constexpr size_t OFF_ST   = OFF_Y + SZ_Y;
constexpr size_t OFF_PR   = OFF_ST + SZ_ST;
constexpr size_t OFF_YN   = OFF_PR + SZ_ST;

extern "C" void kernel_launch(void* const* d_in, const int* in_sizes, int n_in,
                              void* d_out, int out_size, void* d_ws, size_t ws_size,
                              hipStream_t stream) {
  const float* u          = (const float*)d_in[0];
  const float* in_proj_w  = (const float*)d_in[1];
  const float* conv_w     = (const float*)d_in[2];
  const float* conv_b     = (const float*)d_in[3];
  const float* dt_bias    = (const float*)d_in[4];
  const float* A_log      = (const float*)d_in[5];
  const float* D_param    = (const float*)d_in[6];
  const float* norm_w     = (const float*)d_in[7];
  const float* out_proj_w = (const float*)d_in[8];
  const float* out_proj_b = (const float*)d_in[9];
  float* out = (float*)d_out;

  char* ws = (char*)d_ws;
  float*          g_z     = (float*)(ws + OFF_Z);
  float*          g_dtsp  = (float*)(ws + OFF_DTSP);
  float*          g_dA    = (float*)(ws + OFF_DA);
  float*          g_acum  = (float*)(ws + OFF_ACUM);
  float*          g_x     = (float*)(ws + OFF_X);
  unsigned short* g_xdt   = (unsigned short*)(ws + OFF_XDT);
  unsigned short* g_Bbf   = (unsigned short*)(ws + OFF_BBF);
  unsigned short* g_Cbf   = (unsigned short*)(ws + OFF_CBF);
  float*          g_y     = (float*)(ws + OFF_Y);
  float*          g_st    = (float*)(ws + OFF_ST);
  float*          g_pr    = (float*)(ws + OFF_PR);
  float*          g_yn    = (float*)(ws + OFF_YN);

  // 1) in_proj: [4096,1024] x [4368,1024]^T
  k_gemm<<<dim3((kDIP + 63) / 64, kBL / 128), 256, 0, stream>>>(
      u, in_proj_w, nullptr, g_z, kBL, kDIP, kDM);

  // 2) dt softplus + dA
  k_dtprep<<<(kBL * kH) / 256, 256, 0, stream>>>(g_z, dt_bias, A_log, g_dtsp, g_dA);

  // 3) depthwise conv + SiLU + routing
  k_conv<<<(int)(((size_t)kBL * kCONV + 255) / 256), 256, 0, stream>>>(
      g_z, conv_w, conv_b, g_dtsp, g_x, g_xdt, g_Bbf, g_Cbf);

  // 4) per-chunk cumsum of dA
  k_cumsum<<<kB * kNC * kH, 256, 0, stream>>>(g_dA, g_acum);

  // 5) Y_diag (WMMA)
  size_t smem_ydiag = (size_t)(kCS * kN + 64 * kN + 64 * kP + 8 * 32 * 64) * 2 + kCS * 4;
  k_ydiag<<<kB * kNC * kH, 256, smem_ydiag, stream>>>(g_Cbf, g_Bbf, g_xdt, g_acum, g_y);

  // 6) per-chunk states (WMMA)
  size_t smem_states = (size_t)(kCS * kN + kP * kCS) * 2;
  k_states<<<kB * kNC * kH, 256, smem_states, stream>>>(g_Bbf, g_xdt, g_acum, g_st);

  // 7) inter-chunk scan
  k_scan<<<kB * kH, 256, 0, stream>>>(g_st, g_acum, g_pr);

  // 8) Y_off + D*x (WMMA, in-place into g_y)
  size_t smem_yoff = (size_t)(kCS * kN + kP * kN) * 2 + kCS * 4;
  k_yoff<<<kB * kNC * kH, 256, smem_yoff, stream>>>(g_Cbf, g_pr, g_acum, g_x, D_param, g_y);

  // 9) gate + RMS norm
  k_gatenorm<<<kBL, 256, 0, stream>>>(g_y, g_z, norm_w, g_yn);

  // 10) out_proj: [4096,2048] x [1024,2048]^T + bias
  k_gemm<<<dim3(kDM / 64, kBL / 128), 256, 0, stream>>>(
      g_yn, out_proj_w, out_proj_b, out, kBL, kDM, kDI);
}
